// DifferentialMultiHeadAttention_25847113187818
// MI455X (gfx1250) — matrix operand order repaired and checked by reading, not compiled, not run on hardware
//
#include <hip/hip_runtime.h>
#include <math.h>

typedef __bf16 bf16;
typedef __attribute__((ext_vector_type(16))) __bf16 v16bf;
typedef __attribute__((ext_vector_type(8)))  float  v8f;

#define LAMBDA_INIT_F 0.78360576653162453f
#define OUT_SCALE_F   (1.0f - LAMBDA_INIT_F)
#define Q_SCALE_F     0.17677669529663689f   // 32^-0.5
#define SLEN 2048
#define EDIM 1024
#define NHEAD 16

union FragBits { uint4 u[2]; v16bf v; };

__device__ inline v8f vzero() {
    v8f z;
#pragma unroll
    for (int j = 0; j < 8; ++j) z[j] = 0.0f;
    return z;
}

// WMMA bf16: D = A(16x32) * B(32x16) + C, f32 accumulate
__device__ inline v8f wmma_bf16(v16bf a, v16bf b, v8f c) {
    return __builtin_amdgcn_wmma_f32_16x16x32_bf16(false, a, false, b, (short)0, c, false, false);
}

// A-fragment: 16(M) x 32(K) bf16 tile from row-major [*, ld], base p = &A[m0*ld + k0].
// lanes 0-15: M=lane, K 0..7 (u[0]) and 16..23 (u[1]); lanes 16-31: M=lane-16, K 8..15 / 24..31.
__device__ inline v16bf load_a_frag(const bf16* p, int ld) {
    int lane = threadIdx.x & 31;
    int r = lane & 15, hi = lane >> 4;
    const bf16* q = p + (size_t)r * ld + hi * 8;
    FragBits f;
    f.u[0] = *(const uint4*)(q);
    f.u[1] = *(const uint4*)(q + 16);
    return f.v;
}

// B-fragment: 32(K) x 16(N) bf16, source stored as rows of N with contiguous K
// (i.e. Bm[n, k] row-major), base p = &Bm[n0*ld + k0].
// lanes 0-15: N=lane, K 0..7 (u[0]) and 16..23 (u[1]); lanes 16-31: N=lane-16, K 8..15 / 24..31.
__device__ inline v16bf load_b_frag(const bf16* p, int ld) {
    int lane = threadIdx.x & 31;
    int r = lane & 15, hi = lane >> 4;
    const bf16* q = p + (size_t)r * ld + hi * 8;
    FragBits f;
    f.u[0] = *(const uint4*)(q);
    f.u[1] = *(const uint4*)(q + 16);
    return f.v;
}

// ---------------- elementwise f32 -> bf16 convert ----------------
__global__ void cvt_kernel(const float* __restrict__ src, bf16* __restrict__ dst, int n) {
    int i = blockIdx.x * blockDim.x + threadIdx.x;
    int stride = gridDim.x * blockDim.x;
    for (; i < n; i += stride) dst[i] = (bf16)src[i];
}

// ---------------- lambda_full scalar ----------------
__global__ void lambda_kernel(const float* __restrict__ lq1, const float* __restrict__ lk1,
                              const float* __restrict__ lq2, const float* __restrict__ lk2,
                              float* __restrict__ out) {
    int t = threadIdx.x;  // 32 threads == HEAD_DIM
    float a = lq1[t] * lk1[t];
    float b = lq2[t] * lk2[t];
#pragma unroll
    for (int m = 16; m >= 1; m >>= 1) {
        a += __shfl_xor(a, m);
        b += __shfl_xor(b, m);
    }
    if (t == 0) out[0] = __expf(a) - __expf(b) + LAMBDA_INIT_F;
}

// ---------------- GEMM: C[M,N] = alpha * A[M,K] * Bm[N,K]^T ----------------
// MODE 0: f32 output row-major.  MODE 1/2: bf16 output row-major (alpha applied).
// MODE 3: bf16 output transposed per batch: out[(m/2048)*1024*2048 + n*2048 + (m%2048)].
template <int MODE>
__global__ void __launch_bounds__(256)
gemm_bf16(const bf16* __restrict__ A, const bf16* __restrict__ Bm,
          float* __restrict__ Cf, bf16* __restrict__ Cb,
          int M, int N, int K, float alpha) {
    int wid = threadIdx.x >> 5;
    int lane = threadIdx.x & 31;
    int r = lane & 15, hi = lane >> 4;
    int wm = wid & 1, wn = wid >> 1;
    int m0 = blockIdx.y * 128 + wm * 64;
    int n0 = blockIdx.x * 256 + wn * 64;

    v8f c[4][4];
#pragma unroll
    for (int i = 0; i < 4; ++i)
#pragma unroll
        for (int j = 0; j < 4; ++j) c[i][j] = vzero();

    for (int k0 = 0; k0 < K; k0 += 32) {
        v16bf a[4], b[4];
#pragma unroll
        for (int i = 0; i < 4; ++i)
            a[i] = load_a_frag(A + (size_t)(m0 + 16 * i) * K + k0, K);
#pragma unroll
        for (int j = 0; j < 4; ++j)
            b[j] = load_b_frag(Bm + (size_t)(n0 + 16 * j) * K + k0, K);
#pragma unroll
        for (int i = 0; i < 4; ++i)
#pragma unroll
            for (int j = 0; j < 4; ++j)
                c[i][j] = wmma_bf16(a[i], b[j], c[i][j]);
    }

#pragma unroll
    for (int i = 0; i < 4; ++i)
#pragma unroll
        for (int j = 0; j < 4; ++j)
#pragma unroll
            for (int e = 0; e < 8; ++e) {
                int mm = m0 + 16 * i + hi * 8 + e;
                int nn = n0 + 16 * j + r;
                float v = c[i][j][e] * alpha;
                if (MODE == 0) {
                    Cf[(size_t)mm * N + nn] = v;
                } else if (MODE == 3) {
                    Cb[(size_t)(mm >> 11) * ((size_t)EDIM * SLEN) + (size_t)nn * SLEN + (mm & 2047)] = (bf16)v;
                } else {
                    Cb[(size_t)mm * N + nn] = (bf16)v;
                }
            }
}

// ---------------- fused flash differential attention + RMSNorm ----------------
// One wave owns a 16-row q-tile of one (batch, head); streams keys in blocks of 32,
// maintaining two online softmaxes (heads 2h and 2h+1).  Keeps the whole working
// set inside the 256-VGPR direct window (no s_set_vgpr_msb churn).
__device__ inline void flash_update(v8f s0, v8f s1, v8f& m, v8f& l, v8f* acc,
                                    bf16* pb, const v16bf* vf, int hi, int r) {
    // chunk row-max over 32 keys (columns live in lanes; rows in v8f elements)
    v8f cm;
#pragma unroll
    for (int j = 0; j < 8; ++j) cm[j] = fmaxf(s0[j], s1[j]);
#pragma unroll
    for (int d = 8; d >= 1; d >>= 1)
#pragma unroll
        for (int j = 0; j < 8; ++j) cm[j] = fmaxf(cm[j], __shfl_xor(cm[j], d));

    v8f nm, sc, p0, p1, cs;
#pragma unroll
    for (int j = 0; j < 8; ++j) {
        nm[j] = fmaxf(m[j], cm[j]);
        sc[j] = __expf(m[j] - nm[j]);
        p0[j] = __expf(s0[j] - nm[j]);
        p1[j] = __expf(s1[j] - nm[j]);
        cs[j] = p0[j] + p1[j];
    }
#pragma unroll
    for (int d = 8; d >= 1; d >>= 1)
#pragma unroll
        for (int j = 0; j < 8; ++j) cs[j] += __shfl_xor(cs[j], d);
#pragma unroll
    for (int j = 0; j < 8; ++j) {
        l[j] = l[j] * sc[j] + cs[j];
        m[j] = nm[j];
    }
#pragma unroll
    for (int t = 0; t < 4; ++t)
#pragma unroll
        for (int j = 0; j < 8; ++j) acc[t][j] *= sc[j];

    // transpose P (C-layout) -> A-layout via LDS bounce (bf16 16x32 tile)
#pragma unroll
    for (int j = 0; j < 8; ++j) {
        pb[(hi * 8 + j) * 32 + r] = (bf16)p0[j];
        pb[(hi * 8 + j) * 32 + 16 + r] = (bf16)p1[j];
    }
    v16bf pf = load_a_frag(pb, 32);
#pragma unroll
    for (int t = 0; t < 4; ++t) acc[t] = wmma_bf16(pf, vf[t], acc[t]);
}

__global__ void __launch_bounds__(128)
diff_attn(const bf16* __restrict__ Q, const bf16* __restrict__ Kc,
          const bf16* __restrict__ Vt, const float* __restrict__ subw,
          const float* __restrict__ lam_p, bf16* __restrict__ Ab) {
    __shared__ bf16 pbuf[4][16 * 32];
    int wid = threadIdx.x >> 5, lane = threadIdx.x & 31;
    int r = lane & 15, hi = lane >> 4;
    int gt = blockIdx.x * 4 + wid;          // 4096 wave-tiles total
    int qt = gt & 127;                      // S/16 q-tiles
    int h = (gt >> 7) & (NHEAD - 1);
    int b = gt >> 11;
    int q0 = qt * 16;

    const bf16* Qp = Q + (size_t)(b * SLEN + q0) * EDIM + (size_t)(2 * h) * 32;
    v16bf qa = load_a_frag(Qp, EDIM);        // head 2h   (16 x 32)
    v16bf qb = load_a_frag(Qp + 32, EDIM);   // head 2h+1

    v8f m1, l1, m2, l2;
    v8f acc1[4], acc2[4];
#pragma unroll
    for (int j = 0; j < 8; ++j) { m1[j] = -1e30f; m2[j] = -1e30f; l1[j] = 0.f; l2[j] = 0.f; }
#pragma unroll
    for (int t = 0; t < 4; ++t) { acc1[t] = vzero(); acc2[t] = vzero(); }

    float lam = lam_p[0];
    bf16* pb = pbuf[wid];

    for (int kb = 0; kb < SLEN; kb += 32) {
        const bf16* Kp = Kc + (size_t)(b * SLEN + kb) * EDIM + (size_t)(2 * h) * 32;
        v16bf k_a0 = load_b_frag(Kp, EDIM);                           // head 2h, keys kb..kb+15
        v16bf k_a1 = load_b_frag(Kp + (size_t)16 * EDIM, EDIM);       // keys kb+16..kb+31
        v16bf k_b0 = load_b_frag(Kp + 32, EDIM);                      // head 2h+1
        v16bf k_b1 = load_b_frag(Kp + 32 + (size_t)16 * EDIM, EDIM);

        v8f z = vzero();
        v8f s10 = wmma_bf16(qa, k_a0, z);
        v8f s11 = wmma_bf16(qa, k_a1, z);
        v8f s20 = wmma_bf16(qb, k_b0, z);
        v8f s21 = wmma_bf16(qb, k_b1, z);

        v16bf vf[4];   // V block: Vt[b, h*64 + t*16 + lane, kb..kb+31], contiguous in keys
#pragma unroll
        for (int t = 0; t < 4; ++t)
            vf[t] = load_b_frag(Vt + (size_t)(b * EDIM + h * 64 + t * 16) * SLEN + kb, SLEN);

        flash_update(s10, s11, m1, l1, acc1, pb, vf, hi, r);
        flash_update(s20, s21, m2, l2, acc2, pb, vf, hi, r);
    }

    // combine: out = acc1/l1 - lam*acc2/l2, then RMSNorm over 64 cols, scale, store bf16
    v8f i1, i2, o[4];
#pragma unroll
    for (int j = 0; j < 8; ++j) { i1[j] = 1.0f / l1[j]; i2[j] = lam / l2[j]; }
#pragma unroll
    for (int t = 0; t < 4; ++t)
#pragma unroll
        for (int j = 0; j < 8; ++j) o[t][j] = acc1[t][j] * i1[j] - acc2[t][j] * i2[j];

    v8f ss = vzero();
#pragma unroll
    for (int t = 0; t < 4; ++t)
#pragma unroll
        for (int j = 0; j < 8; ++j) ss[j] += o[t][j] * o[t][j];
#pragma unroll
    for (int d = 8; d >= 1; d >>= 1)
#pragma unroll
        for (int j = 0; j < 8; ++j) ss[j] += __shfl_xor(ss[j], d);

    float wcol[4];
#pragma unroll
    for (int t = 0; t < 4; ++t) wcol[t] = subw[t * 16 + r];

#pragma unroll
    for (int t = 0; t < 4; ++t)
#pragma unroll
        for (int j = 0; j < 8; ++j) {
            float rms = rsqrtf(ss[j] * (1.0f / 64.0f) + 1e-5f);
            float v = o[t][j] * rms * wcol[t] * OUT_SCALE_F;
            Ab[(size_t)(b * SLEN + q0 + hi * 8 + j) * EDIM + h * 64 + t * 16 + r] = (bf16)v;
        }
}

// ---------------- launch ----------------
extern "C" void kernel_launch(void* const* d_in, const int* in_sizes, int n_in,
                              void* d_out, int out_size, void* d_ws, size_t ws_size,
                              hipStream_t stream) {
    (void)in_sizes; (void)n_in; (void)out_size; (void)ws_size;
    const float* x   = (const float*)d_in[0];
    const float* Wq  = (const float*)d_in[1];
    const float* Wk  = (const float*)d_in[2];
    const float* Wv  = (const float*)d_in[3];
    const float* Wo  = (const float*)d_in[4];
    const float* lq1 = (const float*)d_in[5];
    const float* lk1 = (const float*)d_in[6];
    const float* lq2 = (const float*)d_in[7];
    const float* lk2 = (const float*)d_in[8];
    const float* sub = (const float*)d_in[9];
    float* out = (float*)d_out;

    const size_t NX = (size_t)2 * SLEN * EDIM;   // 4M elems
    const size_t NW = (size_t)EDIM * EDIM;       // 1M elems
    char* ws = (char*)d_ws;
    float* lam = (float*)ws;                     size_t off = 256;
    bf16* xb  = (bf16*)(ws + off); off += NX * 2;
    bf16* wqb = (bf16*)(ws + off); off += NW * 2;
    bf16* wkb = (bf16*)(ws + off); off += NW * 2;
    bf16* wvb = (bf16*)(ws + off); off += NW * 2;
    bf16* wob = (bf16*)(ws + off); off += NW * 2;
    bf16* qb  = (bf16*)(ws + off); off += NX * 2;
    bf16* kbf = (bf16*)(ws + off); off += NX * 2;
    bf16* vt  = (bf16*)(ws + off); off += NX * 2;
    bf16* ab  = (bf16*)(ws + off); off += NX * 2;

    cvt_kernel<<<512, 256, 0, stream>>>(x, xb, (int)NX);
    cvt_kernel<<<256, 256, 0, stream>>>(Wq, wqb, (int)NW);
    cvt_kernel<<<256, 256, 0, stream>>>(Wk, wkb, (int)NW);
    cvt_kernel<<<256, 256, 0, stream>>>(Wv, wvb, (int)NW);
    cvt_kernel<<<256, 256, 0, stream>>>(Wo, wob, (int)NW);
    lambda_kernel<<<1, 32, 0, stream>>>(lq1, lk1, lq2, lk2, lam);

    dim3 g(EDIM / 256, (2 * SLEN) / 128);  // (4, 32)
    gemm_bf16<1><<<g, 256, 0, stream>>>(xb, wqb, nullptr, qb, 2 * SLEN, EDIM, EDIM, Q_SCALE_F);
    gemm_bf16<2><<<g, 256, 0, stream>>>(xb, wkb, nullptr, kbf, 2 * SLEN, EDIM, EDIM, 1.0f);
    gemm_bf16<3><<<g, 256, 0, stream>>>(xb, wvb, nullptr, vt, 2 * SLEN, EDIM, EDIM, 1.0f);

    diff_attn<<<1024, 128, 0, stream>>>(qb, kbf, vt, sub, lam, ab);

    gemm_bf16<0><<<g, 256, 0, stream>>>(ab, wob, out, nullptr, 2 * SLEN, EDIM, EDIM, 1.0f);
}
